// MultiHeadAttention_81681688036014
// MI455X (gfx1250) — matrix operand order repaired and checked by reading, not compiled, not run on hardware
//
#include <hip/hip_runtime.h>

// =====================================================================
// Causal multi-head attention forward for MI455X (gfx1250, wave32, WMMA)
//
// Shapes: x[2,2048,1024], Wq/Wk/Wv/Wo[1024,1024], bo[1024]; H=16, Dh=64.
// Compute-bound (~68 GFLOP vs ~100MB traffic @ 23.3 TB/s), so all four
// matmul stages use v_wmma_f32_16x16x32_bf16 (fp32 accumulate). Whole
// working set is L2-resident (192MB), so fragment loads come straight
// from global (WGP$/L2) with branchless global_prefetch hints; weights
// and V are pre-transposed so every per-lane fragment load is a
// contiguous b128. Attention uses a 64-keys-per-iteration flash loop so
// the wave32 shfl_xor softmax reductions amortize over more keys.
// =====================================================================

typedef __attribute__((ext_vector_type(16))) __bf16 bf16x16;
typedef __attribute__((ext_vector_type(8)))  __bf16 bf16x8;
typedef __attribute__((ext_vector_type(8)))  float  f32x8;

union BF16x16U { bf16x16 v; bf16x8 h[2]; };

#define B_DIM   2
#define T_DIM   2048
#define D_MODEL 1024
#define NHEAD   16
#define HDIM    64
#define M_ROWS  (B_DIM * T_DIM)        // 4096
#define VT_LD   (T_DIM + 64)           // padded V^T row stride (in elems)

// ---- workspace layout (bytes) ----
#define OFF_XBF  (0ull)
#define OFF_WQT  (OFF_XBF + (size_t)M_ROWS * D_MODEL * 2)
#define OFF_WKT  (OFF_WQT + (size_t)D_MODEL * D_MODEL * 2)
#define OFF_WVT  (OFF_WKT + (size_t)D_MODEL * D_MODEL * 2)
#define OFF_WOT  (OFF_WVT + (size_t)D_MODEL * D_MODEL * 2)
#define OFF_Q    (OFF_WOT + (size_t)D_MODEL * D_MODEL * 2)
#define OFF_K    (OFF_Q   + (size_t)M_ROWS * D_MODEL * 2)
#define OFF_VT   (OFF_K   + (size_t)M_ROWS * D_MODEL * 2)
#define OFF_CTX  (OFF_VT  + (size_t)B_DIM * NHEAD * HDIM * VT_LD * 2)

static __device__ inline __bf16 f2bf(float f) {
  unsigned int u = __builtin_bit_cast(unsigned int, f);
  u += 0x7FFFu + ((u >> 16) & 1u);                   // round-to-nearest-even
  unsigned short hs = (unsigned short)(u >> 16);
  return __builtin_bit_cast(__bf16, hs);
}

// A-fragment (16x32 bf16, M x K): lane<16 -> row m0+l, K k0..k0+15
//                                 lane>=16 -> row m0+l, K k0+16..k0+31
static __device__ inline bf16x16 load_a_frag(const __bf16* base, int ld,
                                             int m0, int k0, int lane) {
  const int l = lane & 15, hi = lane >> 4;
  const __bf16* p = base + (size_t)(m0 + l) * ld + k0 + hi * 16;
  BF16x16U r;
  r.h[0] = *(const bf16x8*)(p);
  r.h[1] = *(const bf16x8*)(p + 8);
  return r.v;
}

// B-fragment (32x16 bf16, K x N) from a K-major (transposed) array Bt[n][k]:
// lane<16 -> col n0+l holds K k0..k0+15 ; lane>=16 -> col n0+l holds K k0+16..k0+31
static __device__ inline bf16x16 load_b_frag(const __bf16* baseT, int ld,
                                             int n0, int k0, int lane) {
  const int l = lane & 15, hi = lane >> 4;
  const __bf16* p = baseT + (size_t)(n0 + l) * ld + k0 + hi * 16;
  BF16x16U r;
  r.h[0] = *(const bf16x8*)(p);
  r.h[1] = *(const bf16x8*)(p + 8);
  return r.v;
}

static __device__ inline f32x8 wmma_bf16(bf16x16 a, bf16x16 b, f32x8 c) {
  return __builtin_amdgcn_wmma_f32_16x16x32_bf16(
      false, a, false, b, (short)0, c, false, false);
}

// ---------------------------------------------------------------------
// 1) fp32 -> bf16 conversion: x row-major, weights transposed (Wt[n][k])
// ---------------------------------------------------------------------
__global__ __launch_bounds__(256) void convert_kernel(
    const float* __restrict__ x,
    const float* __restrict__ Wq, const float* __restrict__ Wk,
    const float* __restrict__ Wv, const float* __restrict__ Wo,
    __bf16* __restrict__ Xbf,
    __bf16* __restrict__ WqT, __bf16* __restrict__ WkT,
    __bf16* __restrict__ WvT, __bf16* __restrict__ WoT) {
  const size_t NX = (size_t)M_ROWS * D_MODEL;               // 4,194,304
  size_t i = (size_t)blockIdx.x * blockDim.x + threadIdx.x;
  if (i < NX) { Xbf[i] = f2bf(x[i]); return; }
  size_t j = i - NX;                                        // 4 x 1M weights
  const int which = (int)(j >> 20);
  const int e = (int)(j & 1048575u);
  const int k = e >> 10, n = e & 1023;
  const float* src = (which == 0) ? Wq : (which == 1) ? Wk : (which == 2) ? Wv : Wo;
  __bf16*      dst = (which == 0) ? WqT : (which == 1) ? WkT : (which == 2) ? WvT : WoT;
  dst[(size_t)n * D_MODEL + k] = f2bf(src[e]);
}

// ---------------------------------------------------------------------
// 2) WMMA GEMM: C[M,N] = A[M,K] * B[K,N], B given K-major (Bt[n][k]).
// Block = 256 threads = 8 waves (4 row-groups x 2 col-groups),
// wave tile = 32x64 (2x4 WMMA accumulators), block tile = 128x128.
// mode 0: bf16 row-major out; mode 1: bf16 V^T out [b,h,d,t] (ld=VT_LD);
// mode 2: fp32 out + bias.
// ---------------------------------------------------------------------
__global__ __launch_bounds__(256) void gemm_bf16_kernel(
    const __bf16* __restrict__ A, const __bf16* __restrict__ Bt,
    void* __restrict__ out, const float* __restrict__ bias,
    int M, int N, int K, int mode) {
  const int lane = threadIdx.x & 31;
  const int w = threadIdx.x >> 5;
  const int wrow = w & 3, wcol = w >> 2;
  const int l = lane & 15, hi = lane >> 4;
  const int rowbase = blockIdx.x * 128 + wrow * 32;
  const int colbase = blockIdx.y * 128 + wcol * 64;

  f32x8 acc[2][4] = {};
  for (int k0 = 0; k0 < K; k0 += 32) {
    // Branchless WGP-scope prefetch of the K-tile two steps ahead (global_prefetch_b8).
    // K is 1024 here; wrapping with &1023 on the final tiles just re-touches
    // the row head, keeping the loop a single basic block so the scheduler
    // can keep the deep clause/partial-wait pipelining.
    {
      const int kp = (k0 + 64) & 1023;
      __builtin_prefetch(A + (size_t)(rowbase + lane) * K + kp, 0, 3);
      __builtin_prefetch(Bt + (size_t)(colbase + lane) * K + kp, 0, 3);
    }
    bf16x16 a0 = load_a_frag(A, K, rowbase, k0, lane);
    bf16x16 a1 = load_a_frag(A, K, rowbase + 16, k0, lane);
#pragma unroll
    for (int ci = 0; ci < 4; ++ci) {
      bf16x16 b = load_b_frag(Bt, K, colbase + 16 * ci, k0, lane);
      acc[0][ci] = wmma_bf16(a0, b, acc[0][ci]);
      acc[1][ci] = wmma_bf16(a1, b, acc[1][ci]);
    }
  }
#pragma unroll
  for (int ri = 0; ri < 2; ++ri)
#pragma unroll
    for (int ci = 0; ci < 4; ++ci)
#pragma unroll
      for (int r = 0; r < 8; ++r) {
        const int m = rowbase + 16 * ri + r + 8 * hi;
        const int n = colbase + 16 * ci + l;
        const float v = acc[ri][ci][r];
        if (mode == 2) {
          ((float*)out)[(size_t)m * N + n] = v + bias[n];
        } else if (mode == 0) {
          ((__bf16*)out)[(size_t)m * N + n] = f2bf(v);
        } else {  // V^T: [b, h, dd, t] with padded ld
          const int bb = m >> 11, t = m & (T_DIM - 1);
          const int hh = n >> 6, dd = n & (HDIM - 1);
          ((__bf16*)out)[((size_t)((bb * NHEAD + hh) * HDIM + dd)) * VT_LD + t] =
              f2bf(v);
        }
      }
}

// ---------------------------------------------------------------------
// 3) Flash attention: one wave per (b,h, 16-query tile); 64 keys / iter
//    (4x 16-key score sub-tiles -> one set of shfl_xor row reductions
//    covers 64 keys). Score WMMAs -> online softmax -> P transposed via
//    per-wave LDS -> ctx WMMAs against pre-transposed V.
// ---------------------------------------------------------------------
__global__ __launch_bounds__(256) void attn_kernel(
    const __bf16* __restrict__ Qb, const __bf16* __restrict__ Kb,
    const __bf16* __restrict__ Vt, __bf16* __restrict__ ctx) {
  __shared__ __bf16 sP[8][16][64];  // 16 KB, one 16x64 P tile per wave
  const int lane = threadIdx.x & 31;
  const int w = threadIdx.x >> 5;
  const int l = lane & 15, hi = lane >> 4;
  const int gid = blockIdx.x * 8 + w;          // 0..4095
  const int bh = gid >> 7;                     // 128 q-tiles per (b,h)
  const int qt = gid & 127;
  const int b = bh >> 4, h = bh & 15;
  const int q0 = qt << 4;

  const __bf16* Qrow = Qb + (size_t)b * T_DIM * D_MODEL + h * HDIM;
  const __bf16* Krow = Kb + (size_t)b * T_DIM * D_MODEL + h * HDIM;
  const __bf16* Vhd  = Vt + (size_t)bh * HDIM * VT_LD;

  // Q tile 16x64, kept in registers as two A-fragments (K-steps 0,32)
  const bf16x16 qa0 = load_a_frag(Qrow, D_MODEL, q0, 0, lane);
  const bf16x16 qa1 = load_a_frag(Qrow, D_MODEL, q0, 32, lane);

  f32x8 acc[4] = {};                 // ctx 16x64 accumulators
  float mi[8], li[8];
#pragma unroll
  for (int r = 0; r < 8; ++r) { mi[r] = -__builtin_inff(); li[r] = 0.f; }

  for (int k0 = 0; k0 <= q0 + 15; k0 += 64) {
    float sv[4][8];
#pragma unroll
    for (int s = 0; s < 4; ++s) {
      const int j0 = k0 + 16 * s;
      if (j0 <= q0 + 15) {           // sub-tile has at least one unmasked key
        bf16x16 kb0 = load_b_frag(Krow, D_MODEL, j0, 0, lane);
        bf16x16 kb1 = load_b_frag(Krow, D_MODEL, j0, 32, lane);
        f32x8 sc = {};
        sc = wmma_bf16(qa0, kb0, sc);
        sc = wmma_bf16(qa1, kb1, sc);
#pragma unroll
        for (int r = 0; r < 8; ++r) {
          const int key = j0 + l;
          const int qq = q0 + r + 8 * hi;
          sv[s][r] = (key <= qq) ? sc[r] * 0.125f : -__builtin_inff();
        }
      } else {
#pragma unroll
        for (int r = 0; r < 8; ++r) sv[s][r] = -__builtin_inff();
      }
    }
    // Online softmax: rows live across the 16-lane half-groups.
    // One shuffle-reduction pass covers 64 keys.
    float scale_[8];
#pragma unroll
    for (int r = 0; r < 8; ++r) {
      float mx = fmaxf(fmaxf(sv[0][r], sv[1][r]), fmaxf(sv[2][r], sv[3][r]));
#pragma unroll
      for (int msk = 1; msk <= 8; msk <<= 1)
        mx = fmaxf(mx, __shfl_xor(mx, msk, 32));
      const float mnew = fmaxf(mi[r], mx);
      const float p0 = __expf(sv[0][r] - mnew);
      const float p1 = __expf(sv[1][r] - mnew);
      const float p2 = __expf(sv[2][r] - mnew);
      const float p3 = __expf(sv[3][r] - mnew);
      float rs = (p0 + p1) + (p2 + p3);
#pragma unroll
      for (int msk = 1; msk <= 8; msk <<= 1)
        rs += __shfl_xor(rs, msk, 32);
      const float f = __expf(mi[r] - mnew);
      li[r] = li[r] * f + rs;
      mi[r] = mnew;
      scale_[r] = f;
      // stash P in C-layout; reload below in A-layout (transpose via LDS)
      const int row = r + 8 * hi;
      sP[w][row][l]      = f2bf(p0);
      sP[w][row][16 + l] = f2bf(p1);
      sP[w][row][32 + l] = f2bf(p2);
      sP[w][row][48 + l] = f2bf(p3);
    }
#pragma unroll
    for (int g = 0; g < 4; ++g)
#pragma unroll
      for (int r = 0; r < 8; ++r) acc[g][r] *= scale_[r];

    // wave-private LDS tile; waves diverge, so no s_barrier — just drain DS
    asm volatile("s_wait_dscnt 0" ::: "memory");
#pragma unroll
    for (int ks = 0; ks < 2; ++ks) {
      // P 16x64 -> two A-fragments (K-steps 0, 32) read back in A layout
      const bf16x16 pf =
          load_a_frag(&sP[w][0][0], 64, 0, 32 * ks, lane);
#pragma unroll
      for (int g = 0; g < 4; ++g) {
        // V^T[d][t]: B-fragment cols = d group, K = 32 keys (contiguous t)
        bf16x16 vb = load_b_frag(Vhd + k0 + 32 * ks, VT_LD, 16 * g, 0, lane);
        acc[g] = wmma_bf16(pf, vb, acc[g]);
      }
    }
  }
  // normalize and merge heads back to [b, t, h*HDIM]
#pragma unroll
  for (int g = 0; g < 4; ++g)
#pragma unroll
    for (int r = 0; r < 8; ++r) {
      const int row = q0 + r + 8 * hi;
      const int col = h * HDIM + 16 * g + l;
      ctx[((size_t)b * T_DIM + row) * D_MODEL + col] = f2bf(acc[g][r] / li[r]);
    }
}

// ---------------------------------------------------------------------
extern "C" void kernel_launch(void* const* d_in, const int* in_sizes, int n_in,
                              void* d_out, int out_size, void* d_ws, size_t ws_size,
                              hipStream_t stream) {
  (void)in_sizes; (void)n_in; (void)out_size; (void)ws_size;
  const float* x  = (const float*)d_in[0];
  const float* Wq = (const float*)d_in[1];
  const float* Wk = (const float*)d_in[2];
  const float* Wv = (const float*)d_in[3];
  const float* Wo = (const float*)d_in[4];
  const float* bo = (const float*)d_in[5];

  char* ws = (char*)d_ws;
  __bf16* Xbf = (__bf16*)(ws + OFF_XBF);
  __bf16* WqT = (__bf16*)(ws + OFF_WQT);
  __bf16* WkT = (__bf16*)(ws + OFF_WKT);
  __bf16* WvT = (__bf16*)(ws + OFF_WVT);
  __bf16* WoT = (__bf16*)(ws + OFF_WOT);
  __bf16* Qb  = (__bf16*)(ws + OFF_Q);
  __bf16* Kb  = (__bf16*)(ws + OFF_K);
  __bf16* Vt  = (__bf16*)(ws + OFF_VT);
  __bf16* ctx = (__bf16*)(ws + OFF_CTX);

  // 1) convert inputs to bf16 (weights transposed for K-major B frags)
  {
    const size_t total = (size_t)M_ROWS * D_MODEL + 4ull * D_MODEL * D_MODEL;
    const int blocks = (int)((total + 255) / 256);
    convert_kernel<<<blocks, 256, 0, stream>>>(x, Wq, Wk, Wv, Wo,
                                               Xbf, WqT, WkT, WvT, WoT);
  }
  // 2) QKV projections (V stored pre-transposed [b,h,d,t])
  dim3 gproj(M_ROWS / 128, D_MODEL / 128);
  gemm_bf16_kernel<<<gproj, 256, 0, stream>>>(Xbf, WqT, Qb, nullptr,
                                              M_ROWS, D_MODEL, D_MODEL, 0);
  gemm_bf16_kernel<<<gproj, 256, 0, stream>>>(Xbf, WkT, Kb, nullptr,
                                              M_ROWS, D_MODEL, D_MODEL, 0);
  gemm_bf16_kernel<<<gproj, 256, 0, stream>>>(Xbf, WvT, Vt, nullptr,
                                              M_ROWS, D_MODEL, D_MODEL, 1);
  // 3) causal flash attention: 4096 waves, 8 per block
  attn_kernel<<<(B_DIM * NHEAD * (T_DIM / 16)) / 8, 256, 0, stream>>>(
      Qb, Kb, Vt, ctx);
  // 4) output projection + bias -> fp32 d_out
  gemm_bf16_kernel<<<gproj, 256, 0, stream>>>(ctx, WoT, (float*)d_out,
                                              bo, M_ROWS, D_MODEL, D_MODEL, 2);
}
